// MultiscaleDeformableAttention_31009663877234
// MI455X (gfx1250) — matrix operand order repaired and checked by reading, not compiled, not run on hardware
//
#include <hip/hip_runtime.h>
#include <hip/hip_bf16.h>

// ---------------- problem constants ----------------
#define NB     2
#define DMODEL 256
#define MH     8
#define LVL    4
#define KPTS   4
#define DVDIM  32
#define LKDIM  16          // LVL*KPTS
#define QN     12240       // S == Q
#define NQ     24480       // NB * QN

typedef _Float16 half4 __attribute__((ext_vector_type(4)));
typedef _Float16 half8 __attribute__((ext_vector_type(8)));
typedef _Float16 v16h  __attribute__((ext_vector_type(16)));
typedef float    v8f   __attribute__((ext_vector_type(8)));
typedef float    f32x4 __attribute__((ext_vector_type(4)));

// =====================================================================
// Tiled GEMM + bias: C[M x NCOLS] = A[M x 256] * B[256 x NCOLS] + bias
// fp32 in memory, f16 WMMA 16x16x32 compute, fp32 accumulate.
// Block: 256 threads = 8 waves in 2(M)x4(N); block tile 32 x 64.
// K = 256 fixed -> 8 fully unrolled K-steps, double-buffered LDS.
// Requires M % 32 == 0 (24480 = 765*32), NCOLS % 64 == 0.
// =====================================================================
template<int NCOLS>
__global__ __launch_bounds__(256)
void msda_gemm_wmma(const float* __restrict__ A,
                    const float* __restrict__ B,
                    const float* __restrict__ bias,
                    float* __restrict__ C)
{
    constexpr int KD = 256;
    __shared__ __align__(16) _Float16 ldsA[2][32 * 32];   // [row][k]
    __shared__ __align__(16) _Float16 ldsB[2][64 * 32];   // [col][k] transposed

    const int tid  = threadIdx.x;
    const int lane = tid & 31;
    const int wid  = tid >> 5;
    const int waveRow = wid >> 2;              // 0..1
    const int waveCol = wid & 3;               // 0..3
    const int rowBase = blockIdx.y * 32;
    const int colBase = blockIdx.x * 64;

    // --- staging assignments (vectorized: one/two b128 loads per thread) ---
    // A tile 32x32 floats = 256 float4; thread t -> row=t>>3, k4=(t&7)*4
    const int aRow = tid >> 3;
    const int aCol = (tid & 7) * 4;
    // B tile 32x64 floats = 512 float4; f = tid + i*256 -> k=f>>4, n4=(f&15)*4
    const int bK0 = tid >> 4,          bN0 = (tid & 15) * 4;
    const int bK1 = (tid + 256) >> 4,  bN1 = bN0;   // (f&15) identical for +256

    const float* gA = A + (size_t)(rowBase + aRow) * KD + aCol;
    const float* gB0 = B + (size_t)bK0 * NCOLS + colBase + bN0;
    const float* gB1 = B + (size_t)bK1 * NCOLS + colBase + bN1;

    f32x4 ra, rb0, rb1;

    // Fragment lane mapping (wave32, 16-bit A 16x32 / B 32x16; contiguous k = 16*(lane>>4)+s for A and B, not the ISA table order)
    const int row16 = lane & 15;
    const int kgA   = (lane >> 4) * 16;
    const int kbB   = (lane >> 4) * 16;
    const int col16 = lane & 15;

    auto loadTiles = [&](int kt) {
        ra  = *(const f32x4*)(gA + kt);
        rb0 = *(const f32x4*)(gB0 + (size_t)kt * NCOLS);
        rb1 = *(const f32x4*)(gB1 + (size_t)kt * NCOLS);
    };
    auto storeTiles = [&](int buf) {
        half4 ha;
        #pragma unroll
        for (int j = 0; j < 4; ++j) ha[j] = (_Float16)ra[j];
        *(half4*)&ldsA[buf][aRow * 32 + aCol] = ha;        // one ds_store_b64
        #pragma unroll
        for (int j = 0; j < 4; ++j)                        // transposed scatter
            ldsB[buf][(bN0 + j) * 32 + bK0] = (_Float16)rb0[j];
        #pragma unroll
        for (int j = 0; j < 4; ++j)
            ldsB[buf][(bN1 + j) * 32 + bK1] = (_Float16)rb1[j];
    };

    loadTiles(0);
    storeTiles(0);
    __syncthreads();

    v8f acc = {};
    #pragma unroll
    for (int i = 0; i < KD / 32; ++i) {
        const int buf = i & 1;
        if (i < KD / 32 - 1) loadTiles((i + 1) * 32);      // overlap next tile

        union { v16h v; half8 h[2]; } af, bf;
        const _Float16* pa = &ldsA[buf][(waveRow * 16 + row16) * 32 + kgA];
        af.h[0] = *(const half8*)(pa);                     // K = kgA..kgA+7
        af.h[1] = *(const half8*)(pa + 8);                 // K = kgA+8..+15
        const _Float16* pb = &ldsB[buf][(waveCol * 16 + col16) * 32 + kbB];
        bf.h[0] = *(const half8*)(pb);                     // K = kbB..kbB+7
        bf.h[1] = *(const half8*)(pb + 8);                 // K = kbB+8..+15

        if (i < KD / 32 - 1) storeTiles(buf ^ 1);          // fill other buffer
        __syncthreads();

        acc = __builtin_amdgcn_wmma_f32_16x16x32_f16(
                  false, af.v, false, bf.v, (short)0, acc, false, false);
    }

    // --- epilogue: bias + store (C/D layout: lanes 0-15 M=r, 16-31 M=r+8) ---
    const int mLocal  = (lane >> 4) * 8;
    const int nGlobal = colBase + waveCol * 16 + col16;
    const float bv = bias[nGlobal];
    #pragma unroll
    for (int r = 0; r < 8; ++r) {
        int mGlobal = rowBase + waveRow * 16 + mLocal + r;
        C[(size_t)mGlobal * NCOLS + nGlobal] = acc[r] + bv;
    }
}

// =====================================================================
// Softmax over LK per (n,q,m) in-place, offsets -> grid coords in-place.
// One thread per (n,q,m).
// =====================================================================
__global__ __launch_bounds__(256)
void msda_softmax_pos(float* __restrict__ logits,     // [NQ,128] -> weights
                      float* __restrict__ offpos,     // [NQ,256] -> grid xy
                      const float* __restrict__ geom) // [NQ,4]
{
    int t = blockIdx.x * 256 + threadIdx.x;     // exactly NQ*MH threads
    int row = t >> 3;
    int m   = t & 7;

    float* lg = logits + (size_t)row * (MH * LKDIM) + m * LKDIM;
    float v[LKDIM];
    float mx = -3.0e38f;
    #pragma unroll
    for (int p = 0; p < LKDIM; ++p) { v[p] = lg[p]; mx = fmaxf(mx, v[p]); }
    float s = 0.f;
    #pragma unroll
    for (int p = 0; p < LKDIM; ++p) { v[p] = __expf(v[p] - mx); s += v[p]; }
    float inv = 1.0f / s;
    #pragma unroll
    for (int p = 0; p < LKDIM; ++p) lg[p] = v[p] * inv;

    const float* g = geom + (size_t)row * 4;
    const float cx = g[0], cy = g[1];
    const float sx = g[2] * 0.125f;             // wh * OFFSET_SCALE/K
    const float sy = g[3] * 0.125f;
    float* op = offpos + (size_t)row * (MH * LKDIM * 2) + m * (LKDIM * 2);
    #pragma unroll
    for (int p = 0; p < LKDIM; ++p) {
        float px = cx + op[2 * p + 0] * sx;
        float py = cy + op[2 * p + 1] * sy;
        op[2 * p + 0] = 2.0f * px - 1.0f;
        op[2 * p + 1] = 2.0f * py - 1.0f;
    }
}

// =====================================================================
// Bilinear gather + attention-weighted sum.
// One wave per (n,q,m), lane = dv channel (DVDIM == 32 == wave32) so
// every corner read is one coalesced 128B transaction per wave.
// =====================================================================
__global__ __launch_bounds__(256)
void msda_sample_attend(const float* __restrict__ vproj,  // [NB*QN, 256]
                        const float* __restrict__ grid,   // [NQ, 256]
                        const float* __restrict__ attnw,  // [NQ, 128]
                        float* __restrict__ out)          // [NQ, 256]
{
    const int w    = blockIdx.x * 8 + (threadIdx.x >> 5); // wave id, < NQ*MH
    const int lane = threadIdx.x & 31;
    const int m    = w & 7;
    const int row  = w >> 3;                              // n*Q + q
    const int n    = row / QN;

    const float* gp = grid  + (size_t)row * 256 + m * 32;
    const float* ap = attnw + (size_t)row * 128 + m * 16;
    const float* vb = vproj + (size_t)n * QN * 256 + (size_t)m * DVDIM + lane;

    const int HWs[LVL]   = {96, 48, 24, 12};
    const int bases[LVL] = {0, 9216, 11520, 12096};

    float acc = 0.f;
    #pragma unroll
    for (int p = 0; p < LKDIM; ++p) {
        const int l = p >> 2;
        const int H = HWs[l], W = HWs[l];
        const int base = bases[l];
        const float gx = gp[2 * p + 0];
        const float gy = gp[2 * p + 1];
        const float aw = ap[p];

        const float x = (gx + 1.0f) * (0.5f * (float)W) - 0.5f;
        const float y = (gy + 1.0f) * (0.5f * (float)H) - 0.5f;
        const float x0f = floorf(x), y0f = floorf(y);
        const float wx = x - x0f, wy = y - y0f;
        const int x0 = (int)x0f, y0 = (int)y0f;

        const int   cxs[4] = {x0, x0 + 1, x0,     x0 + 1};
        const int   cys[4] = {y0, y0,     y0 + 1, y0 + 1};
        const float cws[4] = {(1.f - wx) * (1.f - wy), wx * (1.f - wy),
                              (1.f - wx) * wy,         wx * wy};
        #pragma unroll
        for (int c = 0; c < 4; ++c) {
            const int xi = cxs[c], yi = cys[c];
            if (xi >= 0 && xi < W && yi >= 0 && yi < H) {  // else weight*valid==0
                acc += aw * cws[c] * vb[(size_t)(base + yi * W + xi) * 256];
            }
        }
    }
    out[(size_t)row * 256 + m * 32 + lane] = acc;
}

// =====================================================================
extern "C" void kernel_launch(void* const* d_in, const int* in_sizes, int n_in,
                              void* d_out, int out_size, void* d_ws, size_t ws_size,
                              hipStream_t stream)
{
    const float* queries = (const float*)d_in[0];   // [N,Q,256]
    const float* geom    = (const float*)d_in[1];   // [N,Q,4]
    const float* srcv    = (const float*)d_in[2];   // [N,S,256]
    const float* W_off   = (const float*)d_in[3];   // [256,256]
    const float* b_off   = (const float*)d_in[4];   // [256]
    const float* W_attn  = (const float*)d_in[5];   // [256,128]
    const float* b_attn  = (const float*)d_in[6];   // [128]
    const float* W_val   = (const float*)d_in[7];   // [256,256]
    const float* b_val   = (const float*)d_in[8];   // [256]
    const float* W_out   = (const float*)d_in[9];   // [256,256]
    const float* b_out   = (const float*)d_in[10];  // [256]
    float* out = (float*)d_out;

    float* ws      = (float*)d_ws;
    float* logits  = ws;                                   // NQ*128 (-> attn weights)
    float* offpos  = logits + (size_t)NQ * 128;            // NQ*256 (-> grid coords)
    float* vproj   = offpos + (size_t)NQ * 256;            // NQ*256
    float* sampled = vproj  + (size_t)NQ * 256;            // NQ*256

    const dim3 blk(256);
    const dim3 gAttn(128 / 64, NQ / 32);   // (2, 765)
    const dim3 gWide(256 / 64, NQ / 32);   // (4, 765)

    // input projections
    msda_gemm_wmma<128><<<gAttn, blk, 0, stream>>>(queries, W_attn, b_attn, logits);
    msda_gemm_wmma<256><<<gWide, blk, 0, stream>>>(queries, W_off,  b_off,  offpos);
    msda_gemm_wmma<256><<<gWide, blk, 0, stream>>>(srcv,    W_val,  b_val,  vproj);

    // softmax + sampling-grid computation (NQ*MH threads == 765 blocks exactly)
    msda_softmax_pos<<<(NQ * MH) / 256, blk, 0, stream>>>(logits, offpos, geom);

    // deformable bilinear gather + attention weighting (8 waves / block)
    msda_sample_attend<<<(NQ * MH) / 8, blk, 0, stream>>>(vproj, offpos, logits, sampled);

    // output projection
    msda_gemm_wmma<256><<<gWide, blk, 0, stream>>>(sampled, W_out, b_out, out);
}
